// GatedRecurrentBlock_69741678953031
// MI455X (gfx1250) — matrix operand order repaired and checked by reading, not compiled, not run on hardware
//
#include <hip/hip_runtime.h>
#include <hip/hip_bf16.h>
#include <math.h>

// ---------------------------------------------------------------------------
// GatedRecurrentBlock on MI455X (gfx1250, wave32, WMMA).
// Pipeline: [transpose weights -> f16] [rmsnorm -> f16] [WMMA GEMM1 + gates]
//           [chunked parallel scan]    [WMMA GEMM2 + residual]
// GEMM tiles are double-buffered in LDS and filled with the CDNA5 async
// global->LDS DMA path (GLOBAL_LOAD_ASYNC_TO_LDS_B128 / s_wait_asynccnt).
// ---------------------------------------------------------------------------

typedef __attribute__((ext_vector_type(16))) _Float16 v16h;
typedef __attribute__((ext_vector_type(8)))  _Float16 v8h;
typedef __attribute__((ext_vector_type(8)))  float    v8f;
typedef __attribute__((ext_vector_type(4)))  int      v4i;

#define B_DIM   4
#define T_DIM   4096
#define D_DIM   1024
#define ROWS    (B_DIM * T_DIM)      // 16384
#define C_DIM   1024                 // d_inner
#define NGATE   3

// GEMM tiling: 256 threads = 8 waves; waves arranged 2(M) x 4(N).
#define TM 128
#define TN 64
#define TK 32

#define AS1 __attribute__((address_space(1)))
#define AS3 __attribute__((address_space(3)))

__device__ __forceinline__ float sigf(float x) { return 1.0f / (1.0f + __expf(-x)); }
__device__ __forceinline__ float geluf(float x) { return 0.5f * x * (1.0f + erff(x * 0.70710678118654752f)); }

// ---- CDNA5 async global->LDS copy (16 bytes per lane) ----------------------
__device__ __forceinline__ void async_ld16(const _Float16* g, _Float16* l) {
#if __has_builtin(__builtin_amdgcn_global_load_async_to_lds_b128)
  __builtin_amdgcn_global_load_async_to_lds_b128(
      (AS1 v4i*)g, (AS3 v4i*)l, /*offset=*/0, /*cpol=*/0);
#else
  *(v8h*)l = *(const v8h*)g;   // fallback: synchronous through VGPRs
#endif
}

#if __has_builtin(__builtin_amdgcn_s_wait_asynccnt)
#define WAIT_ASYNC(n) __builtin_amdgcn_s_wait_asynccnt(n)
#else
#define WAIT_ASYNC(n) asm volatile("s_wait_asynccnt %0" :: "i"(n) : "memory")
#endif

// ---------------- weight transpose + cast: W[K][N] -> Wt[N][K] f16 ----------
__global__ __launch_bounds__(256) void k_transpose_cast(
    const float* __restrict__ W, _Float16* __restrict__ Wt, int K, int N) {
  long i = (long)blockIdx.x * 256 + threadIdx.x;
  if (i < (long)K * N) {
    int k = (int)(i / N);
    int n = (int)(i % N);
    Wt[(long)n * K + k] = (_Float16)W[i];
  }
}

// ---------------- RMSNorm (per row of 1024) -> f16 --------------------------
__global__ __launch_bounds__(256) void k_rmsnorm(
    const float* __restrict__ x, const float* __restrict__ w,
    _Float16* __restrict__ xn) {
  __shared__ float red[256];
  long row = blockIdx.x;
  const float* xr = x + row * (long)D_DIM;
  int tid = threadIdx.x;
  float s = 0.0f;
  #pragma unroll
  for (int i = tid; i < D_DIM; i += 256) { float v = xr[i]; s += v * v; }
  red[tid] = s;
  __syncthreads();
  for (int off = 128; off > 0; off >>= 1) {
    if (tid < off) red[tid] += red[tid + off];
    __syncthreads();
  }
  float inv = rsqrtf(red[0] * (1.0f / D_DIM) + 1e-6f);
  #pragma unroll
  for (int i = tid; i < D_DIM; i += 256)
    xn[row * (long)D_DIM + i] = (_Float16)(xr[i] * inv * w[i]);
}

// ---------------- fragment helpers (contiguous k = 16*hi + s, same in A and B) --------------
// A 16x32 (MxK): lane L<16 -> row M=L, K0..15
//                lane L>=16 -> row M=L-16, K16..31
__device__ __forceinline__ v16h load_fragA(const _Float16* As, int m0, int lr, int hi) {
  const v8h* p = (const v8h*)&As[(m0 + lr) * TK + hi * 16];
  v8h a0 = p[0];      // K  hi*16 .. hi*16+7
  v8h a1 = p[1];      // K hi*16+8 ..
  return __builtin_shufflevector(a0, a1, 0,1,2,3,4,5,6,7,8,9,10,11,12,13,14,15);
}
// B 32x16 (KxN), staged n-major in LDS [n][k]:
// lane L<16 -> col N=L, K0..15 ; lane L>=16 -> col N=L-16, K16..31
__device__ __forceinline__ v16h load_fragB(const _Float16* Ws, int n0, int lr, int hi) {
  const v8h* q = (const v8h*)&Ws[(n0 + lr) * TK + hi * 16];
  return __builtin_shufflevector(q[0], q[1], 0,1,2,3,4,5,6,7,8,9,10,11,12,13,14,15);
}

// ---------------- GEMM1: a,u = gates(xn @ W_in) -----------------------------
// Xn: [ROWS][1024] f16, Wt: [3*1024][1024] f16 (row = gate*1024 + channel)
__global__ __launch_bounds__(256) void k_gemm1_gates(
    const _Float16* __restrict__ Xn, const _Float16* __restrict__ Wt,
    const float* __restrict__ lambda_log,
    _Float16* __restrict__ aOut, _Float16* __restrict__ uOut) {
  __shared__ alignas(16) _Float16 As[2][TM * TK];            // 2 x 8 KB
  __shared__ alignas(16) _Float16 Ws[2][NGATE * TN * TK];    // 2 x 12 KB
  const int tid = threadIdx.x;
  const int lane = tid & 31, wave = tid >> 5;
  const int lr = lane & 15, hi = lane >> 4;
  const int waveM = wave & 1, waveN = wave >> 1;
  const long rowBase = (long)blockIdx.x * TM;
  const int c0 = blockIdx.y * TN;

  // per-thread staging coordinates (constant across K-steps)
  const int arow0 = tid >> 2,  acc0 = (tid & 3) * 8;           // As pass 0
  const int arow1 = arow0 + 64;                                // As pass 1
  const int wrow  = tid >> 2,  wcc  = (tid & 3) * 8;           // Ws per gate

  // issue 5 async 16B transfers per thread for K-step k0 into buffer `buf`
  auto issue = [&](int k0, int buf) {
    async_ld16(&Xn[(rowBase + arow0) * D_DIM + k0 + acc0], &As[buf][arow0 * TK + acc0]);
    async_ld16(&Xn[(rowBase + arow1) * D_DIM + k0 + acc0], &As[buf][arow1 * TK + acc0]);
    #pragma unroll
    for (int g = 0; g < NGATE; ++g)
      async_ld16(&Wt[(long)(g * C_DIM + c0 + wrow) * D_DIM + k0 + wcc],
                 &Ws[buf][g * TN * TK + wrow * TK + wcc]);
  };

  v8f accA[4] = {}, accR[4] = {}, accV[4] = {};

  const int NT = D_DIM / TK;          // 32 K-steps
  issue(0, 0);
  for (int t = 0; t < NT; ++t) {
    const int cur = t & 1;
    if (t + 1 < NT) { issue((t + 1) * TK, cur ^ 1); WAIT_ASYNC(5); }
    else            { WAIT_ASYNC(0); }
    __syncthreads();                  // all waves' `cur` tile resident in LDS

    v16h Bf[NGATE];
    #pragma unroll
    for (int g = 0; g < NGATE; ++g)
      Bf[g] = load_fragB(&Ws[cur][g * TN * TK], waveN * 16, lr, hi);
    #pragma unroll
    for (int i = 0; i < 4; ++i) {
      v16h Af = load_fragA(As[cur], waveM * 64 + i * 16, lr, hi);
      accA[i] = __builtin_amdgcn_wmma_f32_16x16x32_f16(false, Af, false, Bf[0], (short)0, accA[i], false, false);
      accR[i] = __builtin_amdgcn_wmma_f32_16x16x32_f16(false, Af, false, Bf[1], (short)0, accR[i], false, false);
      accV[i] = __builtin_amdgcn_wmma_f32_16x16x32_f16(false, Af, false, Bf[2], (short)0, accV[i], false, false);
    }
    __syncthreads();                  // done reading `cur` before it is refilled
  }

  // epilogue: C/D layout -> VGPR j holds (M = j + 8*hi, N = lr)
  const int c = c0 + waveN * 16 + lr;
  const float lamc = sigf(lambda_log[c]);
  #pragma unroll
  for (int i = 0; i < 4; ++i) {
    #pragma unroll
    for (int j = 0; j < 8; ++j) {
      long row = rowBase + waveM * 64 + i * 16 + j + hi * 8;
      float av = sigf(accA[i][j]);
      float rv = sigf(accR[i][j]);
      float vv = geluf(accV[i][j]);
      aOut[row * C_DIM + c] = (_Float16)av;
      uOut[row * C_DIM + c] = (_Float16)((1.0f - av) * rv * vv * lamc);
    }
  }
}

// ---------------- chunked linear scan h_t = a_t h_{t-1} + u_t ---------------
#define NCH 32
#define CHL (T_DIM / NCH)   // 128

__global__ __launch_bounds__(256) void k_scan_summ(
    const _Float16* __restrict__ aB, const _Float16* __restrict__ uB,
    float* __restrict__ Ach, float* __restrict__ Uch) {
  long id = (long)blockIdx.x * 256 + threadIdx.x;   // B*NCH*C = 131072
  int c = (int)(id & (C_DIM - 1));
  int r = (int)(id >> 10);
  int chunk = r & (NCH - 1);
  int b = r >> 5;
  float A = 1.0f, U = 0.0f;
  long base = ((long)b * T_DIM + (long)chunk * CHL) * C_DIM + c;
  for (int t = 0; t < CHL; ++t) {
    float af = (float)aB[base + (long)t * C_DIM];
    float uf = (float)uB[base + (long)t * C_DIM];
    U = af * U + uf;
    A *= af;
  }
  long off = ((long)(b * NCH + chunk)) * C_DIM + c;
  Ach[off] = A;
  Uch[off] = U;
}

__global__ __launch_bounds__(256) void k_scan_carry(
    const float* __restrict__ Ach, const float* __restrict__ Uch,
    float* __restrict__ hcar) {
  long id = (long)blockIdx.x * 256 + threadIdx.x;   // B*C = 4096
  int c = (int)(id & (C_DIM - 1));
  int b = (int)(id >> 10);
  float h = 0.0f;
  for (int ch = 0; ch < NCH; ++ch) {
    long off = ((long)(b * NCH + ch)) * C_DIM + c;
    hcar[off] = h;
    h = Ach[off] * h + Uch[off];
  }
}

__global__ __launch_bounds__(256) void k_scan_apply(
    const _Float16* __restrict__ aB, const _Float16* __restrict__ uB,
    const float* __restrict__ hcar, _Float16* __restrict__ yB) {
  long id = (long)blockIdx.x * 256 + threadIdx.x;
  int c = (int)(id & (C_DIM - 1));
  int r = (int)(id >> 10);
  int chunk = r & (NCH - 1);
  int b = r >> 5;
  float h = hcar[((long)(b * NCH + chunk)) * C_DIM + c];
  long base = ((long)b * T_DIM + (long)chunk * CHL) * C_DIM + c;
  for (int t = 0; t < CHL; ++t) {
    float af = (float)aB[base + (long)t * C_DIM];
    float uf = (float)uB[base + (long)t * C_DIM];
    h = af * h + uf;
    yB[base + (long)t * C_DIM] = (_Float16)h;
  }
}

// ---------------- GEMM2: out = residual + y @ W_out -------------------------
__global__ __launch_bounds__(256) void k_gemm2_res(
    const _Float16* __restrict__ Y, const _Float16* __restrict__ WtO,
    const float* __restrict__ xres, float* __restrict__ out) {
  __shared__ alignas(16) _Float16 As[2][TM * TK];
  __shared__ alignas(16) _Float16 Ws[2][TN * TK];
  const int tid = threadIdx.x;
  const int lane = tid & 31, wave = tid >> 5;
  const int lr = lane & 15, hi = lane >> 4;
  const int waveM = wave & 1, waveN = wave >> 1;
  const long rowBase = (long)blockIdx.x * TM;
  const int c0 = blockIdx.y * TN;

  const int arow0 = tid >> 2, acc0 = (tid & 3) * 8;
  const int arow1 = arow0 + 64;
  const int wrow  = tid >> 2, wcc  = (tid & 3) * 8;

  auto issue = [&](int k0, int buf) {
    async_ld16(&Y[(rowBase + arow0) * C_DIM + k0 + acc0], &As[buf][arow0 * TK + acc0]);
    async_ld16(&Y[(rowBase + arow1) * C_DIM + k0 + acc0], &As[buf][arow1 * TK + acc0]);
    async_ld16(&WtO[(long)(c0 + wrow) * C_DIM + k0 + wcc], &Ws[buf][wrow * TK + wcc]);
  };

  v8f acc[4] = {};

  const int NT = C_DIM / TK;
  issue(0, 0);
  for (int t = 0; t < NT; ++t) {
    const int cur = t & 1;
    if (t + 1 < NT) { issue((t + 1) * TK, cur ^ 1); WAIT_ASYNC(3); }
    else            { WAIT_ASYNC(0); }
    __syncthreads();

    v16h Bf = load_fragB(Ws[cur], waveN * 16, lr, hi);
    #pragma unroll
    for (int i = 0; i < 4; ++i) {
      v16h Af = load_fragA(As[cur], waveM * 64 + i * 16, lr, hi);
      acc[i] = __builtin_amdgcn_wmma_f32_16x16x32_f16(false, Af, false, Bf, (short)0, acc[i], false, false);
    }
    __syncthreads();
  }

  const int c = c0 + waveN * 16 + lr;
  #pragma unroll
  for (int i = 0; i < 4; ++i) {
    #pragma unroll
    for (int j = 0; j < 8; ++j) {
      long row = rowBase + waveM * 64 + i * 16 + j + hi * 8;
      out[row * D_DIM + c] = acc[i][j] + xres[row * D_DIM + c];
    }
  }
}

// ---------------------------------------------------------------------------
extern "C" void kernel_launch(void* const* d_in, const int* in_sizes, int n_in,
                              void* d_out, int out_size, void* d_ws, size_t ws_size,
                              hipStream_t stream) {
  const float* x          = (const float*)d_in[0];   // (4,4096,1024)
  const float* w_norm     = (const float*)d_in[1];   // (1024)
  const float* W_in       = (const float*)d_in[2];   // (1024,3072)
  const float* lambda_log = (const float*)d_in[3];   // (1024)
  const float* W_out      = (const float*)d_in[4];   // (1024,1024)
  float* out = (float*)d_out;

  char* w = (char*)d_ws;
  _Float16* Xn   = (_Float16*)w;  w += (size_t)ROWS * D_DIM * 2;           // 32 MB
  _Float16* WtIn = (_Float16*)w;  w += (size_t)NGATE * C_DIM * D_DIM * 2;  // 6 MB
  _Float16* WtOut= (_Float16*)w;  w += (size_t)C_DIM * D_DIM * 2;          // 2 MB
  _Float16* aBuf = (_Float16*)w;  w += (size_t)ROWS * C_DIM * 2;           // 32 MB
  _Float16* uBuf = (_Float16*)w;  w += (size_t)ROWS * C_DIM * 2;           // 32 MB
  _Float16* yBuf = (_Float16*)w;  w += (size_t)ROWS * C_DIM * 2;           // 32 MB
  float* Ach  = (float*)w;        w += (size_t)B_DIM * NCH * C_DIM * 4;
  float* Uch  = (float*)w;        w += (size_t)B_DIM * NCH * C_DIM * 4;
  float* hcar = (float*)w;        w += (size_t)B_DIM * NCH * C_DIM * 4;

  // weights -> f16 transposed [n][k]
  k_transpose_cast<<<(D_DIM * NGATE * C_DIM + 255) / 256, 256, 0, stream>>>(
      W_in, WtIn, D_DIM, NGATE * C_DIM);
  k_transpose_cast<<<(C_DIM * D_DIM + 255) / 256, 256, 0, stream>>>(
      W_out, WtOut, C_DIM, D_DIM);

  // rmsnorm
  k_rmsnorm<<<ROWS, 256, 0, stream>>>(x, w_norm, Xn);

  // fused GEMM1 + gating
  k_gemm1_gates<<<dim3(ROWS / TM, C_DIM / TN), 256, 0, stream>>>(
      Xn, WtIn, lambda_log, aBuf, uBuf);

  // chunked scan
  k_scan_summ<<<(B_DIM * NCH * C_DIM) / 256, 256, 0, stream>>>(aBuf, uBuf, Ach, Uch);
  k_scan_carry<<<(B_DIM * C_DIM) / 256, 256, 0, stream>>>(Ach, Uch, hcar);
  k_scan_apply<<<(B_DIM * NCH * C_DIM) / 256, 256, 0, stream>>>(aBuf, uBuf, hcar, yBuf);

  // GEMM2 + residual
  k_gemm2_res<<<dim3(ROWS / TM, C_DIM / TN), 256, 0, stream>>>(yBuf, WtOut, x, out);
}
